// SpatialCorrelation_38293928411391
// MI455X (gfx1250) — matrix operand order repaired and checked by reading, not compiled, not run on hardware
//
#include <hip/hip_runtime.h>
#include <math.h>

// ---------------- problem constants (match reference) ----------------
#define B_    16
#define N_    48
#define L_    168
#define F_    16
#define H_    4
#define FF_   64
#define DIN_  64
#define DLOC_ 32
#define DE_   32
#define DG_   64
#define G_    8
#define EPS_  1e-5f
#define BN_   (B_ * N_)       // 768
#define LF_   (L_ * F_)       // 2688
#define EG_   (G_ * (G_ - 1)) // 56
#define DIL_  (DIN_ + DLOC_)  // 96

typedef __attribute__((ext_vector_type(16))) _Float16 v16h;
typedef __attribute__((ext_vector_type(8)))  float    v8f;

// =====================================================================
// Kernel 1: per-station transformer encoder layer (post-norm, eval).
// One block per station (B*N = 768 blocks), 256 threads, LDS staging.
// =====================================================================
__global__ void tfm_kernel(const float* __restrict__ x,
                           const float* __restrict__ wqkv, const float* __restrict__ bqkv,
                           const float* __restrict__ wo,   const float* __restrict__ bo,
                           const float* __restrict__ ln1g, const float* __restrict__ ln1b,
                           const float* __restrict__ w1,   const float* __restrict__ b1,
                           const float* __restrict__ w2,   const float* __restrict__ b2,
                           const float* __restrict__ ln2g, const float* __restrict__ ln2b,
                           float* __restrict__ xs_out) {
  __shared__ float s_x[L_][F_];
  __shared__ float s_q[L_][F_];
  __shared__ float s_k[L_][F_];
  __shared__ float s_v[L_][F_];
  __shared__ float s_o[L_][F_];
  const int s   = blockIdx.x;
  const int tid = threadIdx.x;
  const float* xp = x + (size_t)s * L_ * F_;
  for (int i = tid; i < L_ * F_; i += blockDim.x) s_x[i / F_][i % F_] = xp[i];
  __syncthreads();
  // QKV projection: (168x16) @ (16x48)
  for (int i = tid; i < L_ * 3 * F_; i += blockDim.x) {
    int l = i / (3 * F_), c = i % (3 * F_);
    float acc = bqkv[c];
    for (int f = 0; f < F_; ++f) acc += s_x[l][f] * wqkv[f * 3 * F_ + c];
    if (c < F_)            s_q[l][c]          = acc;
    else if (c < 2 * F_)   s_k[l][c - F_]     = acc;
    else                   s_v[l][c - 2 * F_] = acc;
  }
  __syncthreads();
  // multi-head attention, dh = 4, online softmax
  const int dh = F_ / H_;
  for (int i = tid; i < L_ * H_; i += blockDim.x) {
    int l = i / H_, h = i % H_;
    float qv[4];
    for (int d = 0; d < dh; ++d) qv[d] = s_q[l][h * dh + d];
    float m = -1e30f, den = 0.f, acc[4] = {0.f, 0.f, 0.f, 0.f};
    for (int mk = 0; mk < L_; ++mk) {
      float sc = 0.f;
      for (int d = 0; d < dh; ++d) sc += qv[d] * s_k[mk][h * dh + d];
      sc *= 0.5f; // 1/sqrt(dh)
      float nm = fmaxf(m, sc);
      float corr = __expf(m - nm);
      float p = __expf(sc - nm);
      den = den * corr + p;
      for (int d = 0; d < dh; ++d) acc[d] = acc[d] * corr + p * s_v[mk][h * dh + d];
      m = nm;
    }
    float inv = 1.f / den;
    for (int d = 0; d < dh; ++d) s_o[l][h * dh + d] = acc[d] * inv;
  }
  __syncthreads();
  // per-row: out-proj + residual + LN1, FFN + residual + LN2
  for (int l = tid; l < L_; l += blockDim.x) {
    float t[F_];
    for (int f = 0; f < F_; ++f) {
      float acc = bo[f];
      for (int g = 0; g < F_; ++g) acc += s_o[l][g] * wo[g * F_ + f];
      t[f] = s_x[l][f] + acc;
    }
    float mu = 0.f;
    for (int f = 0; f < F_; ++f) mu += t[f];
    mu *= (1.f / F_);
    float var = 0.f;
    for (int f = 0; f < F_; ++f) { float d = t[f] - mu; var += d * d; }
    var *= (1.f / F_);
    float inv = rsqrtf(var + EPS_);
    float x1[F_];
    for (int f = 0; f < F_; ++f) x1[f] = (t[f] - mu) * inv * ln1g[f] + ln1b[f];
    float hb[FF_];
#pragma unroll
    for (int j = 0; j < FF_; ++j) {
      float acc = b1[j];
      for (int f = 0; f < F_; ++f) acc += x1[f] * w1[f * FF_ + j];
      hb[j] = fmaxf(acc, 0.f);
    }
    float y[F_];
    for (int f = 0; f < F_; ++f) {
      float acc = b2[f];
      for (int j = 0; j < FF_; ++j) acc += hb[j] * w2[j * F_ + f];
      y[f] = x1[f] + acc;
    }
    mu = 0.f;
    for (int f = 0; f < F_; ++f) mu += y[f];
    mu *= (1.f / F_);
    var = 0.f;
    for (int f = 0; f < F_; ++f) { float d = y[f] - mu; var += d * d; }
    var *= (1.f / F_);
    inv = rsqrtf(var + EPS_);
    float* op = xs_out + (size_t)s * L_ * F_ + l * F_;
    for (int f = 0; f < F_; ++f) op[f] = (y[f] - mu) * inv * ln2g[f] + ln2b[f];
  }
}

// =====================================================================
// B-matrix fragment packer: f32 (K x 64) row-major -> f16 WMMA fragment
// order: Bp[kstep][tn][lane][e], so the GEMM loads a lane's 16 halves as
// one contiguous 32-byte vector. B packed in contiguous k order (same as A): lanes0-15 K=0..15,
// lanes16-31 K=16..31, two consecutive K per VGPR.
// =====================================================================
__global__ void packB_kernel(const float* __restrict__ Bm, int K,
                             _Float16* __restrict__ Bp) {
  int i = blockIdx.x * blockDim.x + threadIdx.x;   // one per (kstep, tn, lane)
  int total = (K >> 5) * 4 * 32;
  if (i >= total) return;
  int lane = i & 31;
  int tn   = (i >> 5) & 3;
  int ks   = i >> 7;
  int half = lane >> 4, r = lane & 15;
  int n = tn * 16 + r;
  int kbase = (ks << 5) + 16 * half;
  _Float16* dst = Bp + (size_t)i * 16;
#pragma unroll
  for (int e = 0; e < 16; ++e)
    dst[e] = (_Float16)Bm[(size_t)(kbase + e) * 64 + n];
}

// =====================================================================
// WMMA GEMM, M=768, N=64 fixed: C = (relu)(A[768,K] @ B[K,64] + bias).
// One wave per 16x64 output strip (4 accumulators, 4x A-reuse).
// A fragment = two contiguous 8-float runs -> 4x float4 loads + cvt.
// B fragment = packed f16, one 32B vector load per tile.
// No bounds checks: all dims are exact tile multiples.
// =====================================================================
__global__ void __launch_bounds__(256)
wmma_gemm_kernel(const float* __restrict__ A, const _Float16* __restrict__ Bp,
                 const float* __restrict__ bias, float* __restrict__ C,
                 int K, int lda, int relu) {
  const int lane = threadIdx.x;                       // 0..31
  const int tm   = blockIdx.x * blockDim.y + threadIdx.y; // M-tile
  const int half = lane >> 4;
  const int r    = lane & 15;
  const float* arow = A + (size_t)(tm * 16 + r) * lda;
  v8f acc0 = {}, acc1 = {}, acc2 = {}, acc3 = {};
  for (int k0 = 0; k0 < K; k0 += 32) {
    // A 16x32 fragment for this lane: runs [k0+16h .. +7] and [k0+16h+8 .. +7]
    const float4* p1 = (const float4*)(arow + k0 + 16 * half);
    const float4* p2 = (const float4*)(arow + k0 + 16 * half + 8);
    float4 a0 = p1[0], a1 = p1[1], a2 = p2[0], a3 = p2[1];
    v16h a;
    a[0]  = (_Float16)a0.x; a[1]  = (_Float16)a0.y;
    a[2]  = (_Float16)a0.z; a[3]  = (_Float16)a0.w;
    a[4]  = (_Float16)a1.x; a[5]  = (_Float16)a1.y;
    a[6]  = (_Float16)a1.z; a[7]  = (_Float16)a1.w;
    a[8]  = (_Float16)a2.x; a[9]  = (_Float16)a2.y;
    a[10] = (_Float16)a2.z; a[11] = (_Float16)a2.w;
    a[12] = (_Float16)a3.x; a[13] = (_Float16)a3.y;
    a[14] = (_Float16)a3.z; a[15] = (_Float16)a3.w;
    const _Float16* bp = Bp + (((size_t)(k0 >> 5) * 4) * 32 + lane) * 16;
    v16h b0 = *(const v16h*)(bp);
    v16h b1 = *(const v16h*)(bp + 32 * 16);
    v16h b2 = *(const v16h*)(bp + 2 * 32 * 16);
    v16h b3 = *(const v16h*)(bp + 3 * 32 * 16);
    acc0 = __builtin_amdgcn_wmma_f32_16x16x32_f16(false, a, false, b0, (short)0, acc0, false, false);
    acc1 = __builtin_amdgcn_wmma_f32_16x16x32_f16(false, a, false, b1, (short)0, acc1, false, false);
    acc2 = __builtin_amdgcn_wmma_f32_16x16x32_f16(false, a, false, b2, (short)0, acc2, false, false);
    acc3 = __builtin_amdgcn_wmma_f32_16x16x32_f16(false, a, false, b3, (short)0, acc3, false, false);
  }
  const float bs0 = bias ? bias[r]      : 0.f;
  const float bs1 = bias ? bias[16 + r] : 0.f;
  const float bs2 = bias ? bias[32 + r] : 0.f;
  const float bs3 = bias ? bias[48 + r] : 0.f;
#pragma unroll
  for (int rr = 0; rr < 8; ++rr) {
    const int m = tm * 16 + rr + 8 * half;  // C/D: VGPR rr holds M=rr (+8 upper half)
    float* crow = C + (size_t)m * 64;
    float v0 = acc0[rr] + bs0, v1 = acc1[rr] + bs1;
    float v2 = acc2[rr] + bs2, v3 = acc3[rr] + bs3;
    if (relu) {
      v0 = fmaxf(v0, 0.f); v1 = fmaxf(v1, 0.f);
      v2 = fmaxf(v2, 0.f); v3 = fmaxf(v3, 0.f);
    }
    crow[r] = v0; crow[16 + r] = v1; crow[32 + r] = v2; crow[48 + r] = v3;
  }
}

// ---------------- small VALU kernels ----------------
__global__ void loc_grp_kernel(const float* __restrict__ src_locs,
                               const float* __restrict__ wloc, const float* __restrict__ bloc,
                               const float* __restrict__ wgrp, const float* __restrict__ bgrp,
                               float* __restrict__ loce, float* __restrict__ grw) {
  int i = blockIdx.x * blockDim.x + threadIdx.x;
  if (i >= BN_) return;
  float lx = src_locs[i * 2], ly = src_locs[i * 2 + 1];
  for (int d = 0; d < DLOC_; ++d)
    loce[i * DLOC_ + d] = lx * wloc[d] + ly * wloc[DLOC_ + d] + bloc[d];
  float g[G_];
  float mx = -1e30f;
  for (int d = 0; d < G_; ++d) {
    g[d] = lx * wgrp[d] + ly * wgrp[G_ + d] + bgrp[d];
    mx = fmaxf(mx, g[d]);
  }
  float s = 0.f;
  for (int d = 0; d < G_; ++d) { g[d] = __expf(g[d] - mx); s += g[d]; }
  float inv = 1.f / s;
  for (int d = 0; d < G_; ++d) grw[i * G_ + d] = g[d] * inv;
}

// gf[b,g,d] = sum_n grw[b,n,g] * in_locs[b,n,d]; in_locs = [xe | loce]
__global__ void gf_kernel(const float* __restrict__ grw, const float* __restrict__ xe,
                          const float* __restrict__ loce, float* __restrict__ gf) {
  int i = blockIdx.x * blockDim.x + threadIdx.x;
  if (i >= B_ * G_ * DIL_) return;
  int d = i % DIL_, g = (i / DIL_) % G_, b = i / (DIL_ * G_);
  float acc = 0.f;
  for (int n = 0; n < N_; ++n) {
    int st = b * N_ + n;
    float v = (d < DIN_) ? xe[st * DIN_ + d] : loce[st * DLOC_ + (d - DIN_)];
    acc += grw[st * G_ + g] * v;
  }
  gf[i] = acc;
}

// edge attributes: relu([gf_src, gf_dst] @ wedge + bedge)
__global__ void eattr_kernel(const float* __restrict__ gf, const float* __restrict__ wedge,
                             const float* __restrict__ bedge, float* __restrict__ eattr) {
  int i = blockIdx.x * blockDim.x + threadIdx.x;
  if (i >= B_ * EG_ * DE_) return;
  int c = i % DE_, e = (i / DE_) % EG_, b = i / (DE_ * EG_);
  int s = e / (G_ - 1), rem = e % (G_ - 1);
  int d = rem + (rem >= s ? 1 : 0);
  const float* gs = gf + (size_t)(b * G_ + s) * DIL_;
  const float* gd = gf + (size_t)(b * G_ + d) * DIL_;
  float acc = bedge[c];
  for (int k = 0; k < DIL_; ++k) acc += gs[k] * wedge[k * DE_ + c];
  for (int k = 0; k < DIL_; ++k) acc += gd[k] * wedge[(DIL_ + k) * DE_ + c];
  eattr[i] = fmaxf(acc, 0.f);
}

// group-graph message: relu([h_src(HD), eattr(DE)] @ m1w + m1b)
__global__ void gmsg_kernel(const float* __restrict__ h, int HD,
                            const float* __restrict__ eattr,
                            const float* __restrict__ m1w, const float* __restrict__ m1b,
                            float* __restrict__ msg) {
  int i = blockIdx.x * blockDim.x + threadIdx.x;
  if (i >= B_ * EG_ * DG_) return;
  int c = i % DG_, e = (i / DG_) % EG_, b = i / (DG_ * EG_);
  int s = e / (G_ - 1);
  const float* hs = h + (size_t)(b * G_ + s) * HD;
  const float* ea = eattr + (size_t)(b * EG_ + e) * DE_;
  float acc = m1b[c];
  for (int k = 0; k < HD; ++k) acc += hs[k] * m1w[k * DG_ + c];
  for (int k = 0; k < DE_; ++k) acc += ea[k] * m1w[(HD + k) * DG_ + c];
  msg[i] = fmaxf(acc, 0.f);
}

// group-graph scatter-mean over the complete graph on G nodes
__global__ void gagg_kernel(const float* __restrict__ msg, float* __restrict__ agg) {
  int i = blockIdx.x * blockDim.x + threadIdx.x;
  if (i >= B_ * G_ * DG_) return;
  int c = i % DG_, g = (i / DG_) % G_, b = i / (DG_ * G_);
  float acc = 0.f;
  for (int s = 0; s < G_; ++s) {
    if (s == g) continue;
    int e = s * (G_ - 1) + (g - (g > s ? 1 : 0));
    acc += msg[(size_t)(b * EG_ + e) * DG_ + c];
  }
  agg[i] = acc * (1.f / (float)(G_ - 1));
}

// group-graph node update: relu([h(HD), agg(DG)] @ m2w + m2b)
__global__ void gupd_kernel(const float* __restrict__ h, int HD,
                            const float* __restrict__ agg,
                            const float* __restrict__ m2w, const float* __restrict__ m2b,
                            float* __restrict__ out) {
  int i = blockIdx.x * blockDim.x + threadIdx.x;
  if (i >= B_ * G_ * DG_) return;
  int c = i % DG_, node = i / DG_;
  const float* hp = h + (size_t)node * HD;
  const float* ap = agg + (size_t)node * DG_;
  float acc = m2b[c];
  for (int k = 0; k < HD; ++k) acc += hp[k] * m2w[k * DG_ + c];
  for (int k = 0; k < DG_; ++k) acc += ap[k] * m2w[(HD + k) * DG_ + c];
  out[i] = fmaxf(acc, 0.f);
}

// new_x = [xe | grw @ gf2]  (B*N x 128)
__global__ void newx_kernel(const float* __restrict__ xe, const float* __restrict__ grw,
                            const float* __restrict__ gf2, float* __restrict__ newx) {
  const int DX = DIN_ + DG_;
  int i = blockIdx.x * blockDim.x + threadIdx.x;
  if (i >= BN_ * DX) return;
  int d = i % DX, st = i / DX;
  if (d < DIN_) { newx[i] = xe[st * DIN_ + d]; return; }
  int b = st / N_;
  int dd = d - DIN_;
  float acc = 0.f;
  for (int g = 0; g < G_; ++g) acc += grw[st * G_ + g] * gf2[(size_t)(b * G_ + g) * DG_ + dd];
  newx[i] = acc;
}

// station-graph aggregation with analytic edge messages:
// agg[b,j,c] = mean_{i!=j} relu(hW[b,i,c] + dist(i,j)*wlast[c] + m1b[c])
__global__ void nagg_kernel(const float* __restrict__ hW, const float* __restrict__ wlast,
                            const float* __restrict__ m1b, const float* __restrict__ src_locs,
                            float* __restrict__ agg) {
  int i = blockIdx.x * blockDim.x + threadIdx.x;
  if (i >= BN_ * DG_) return;
  int c = i % DG_, j = (i / DG_) % N_, b = i / (DG_ * N_);
  float jx = src_locs[(b * N_ + j) * 2], jy = src_locs[(b * N_ + j) * 2 + 1];
  float wl = wlast[c], bb = m1b[c];
  float acc = 0.f;
  for (int s = 0; s < N_; ++s) {
    if (s == j) continue;
    float dx = src_locs[(b * N_ + s) * 2] - jx;
    float dy = src_locs[(b * N_ + s) * 2 + 1] - jy;
    float dist = sqrtf(dx * dx + dy * dy);
    float m = hW[(size_t)(b * N_ + s) * DG_ + c] + dist * wl + bb;
    acc += fmaxf(m, 0.f);
  }
  agg[i] = acc * (1.f / (float)(N_ - 1));
}

// out = [a(da) | bsrc(db)] row-wise, rows = B*N
__global__ void concat_kernel(const float* __restrict__ a, int da,
                              const float* __restrict__ bsrc, int db, float* __restrict__ out) {
  int dt = da + db;
  int i = blockIdx.x * blockDim.x + threadIdx.x;
  if (i >= BN_ * dt) return;
  int d = i % dt, row = i / dt;
  out[i] = (d < da) ? a[(size_t)row * da + d] : bsrc[(size_t)row * db + (d - da)];
}

// ---------------- host launch ----------------
static inline int cdiv(int a, int b) { return (a + b - 1) / b; }

static void launch_pack(const float* Bm, int K, _Float16* Bp, hipStream_t st) {
  int total = (K >> 5) * 4 * 32;
  packB_kernel<<<cdiv(total, 256), 256, 0, st>>>(Bm, K, Bp);
}

static void launch_wmma(const float* A, const _Float16* Bp, const float* bias, float* C,
                        int K, int lda, int relu, hipStream_t st) {
  // M = 768 -> 48 tiles, 8 waves/block -> 6 blocks; N = 64 fixed.
  wmma_gemm_kernel<<<dim3(6), dim3(32, 8), 0, st>>>(A, Bp, bias, C, K, lda, relu);
}

extern "C" void kernel_launch(void* const* d_in, const int* in_sizes, int n_in,
                              void* d_out, int out_size, void* d_ws, size_t ws_size,
                              hipStream_t stream) {
  (void)in_sizes; (void)n_in; (void)out_size; (void)ws_size;
  const float* x        = (const float*)d_in[0];
  const float* src_locs = (const float*)d_in[1];
  // d_in[2]=tar_loc, d_in[3]=time : unused by this stage
  const float* wqkv = (const float*)d_in[4];
  const float* bqkv = (const float*)d_in[5];
  const float* wo   = (const float*)d_in[6];
  const float* bo   = (const float*)d_in[7];
  const float* ln1g = (const float*)d_in[8];
  const float* ln1b = (const float*)d_in[9];
  const float* w1   = (const float*)d_in[10];
  const float* b1   = (const float*)d_in[11];
  const float* w2   = (const float*)d_in[12];
  const float* b2   = (const float*)d_in[13];
  const float* ln2g = (const float*)d_in[14];
  const float* ln2b = (const float*)d_in[15];
  const float* win  = (const float*)d_in[16];
  const float* b_in = (const float*)d_in[17];
  const float* wloc = (const float*)d_in[18];
  const float* bloc = (const float*)d_in[19];
  const float* wgrp = (const float*)d_in[20];
  const float* bgrp = (const float*)d_in[21];
  const float* wedge = (const float*)d_in[22];
  const float* bedge = (const float*)d_in[23];
  const float* g0m1w = (const float*)d_in[24];
  const float* g0m1b = (const float*)d_in[25];
  const float* g0m2w = (const float*)d_in[26];
  const float* g0m2b = (const float*)d_in[27];
  const float* g1m1w = (const float*)d_in[28];
  const float* g1m1b = (const float*)d_in[29];
  const float* g1m2w = (const float*)d_in[30];
  const float* g1m2b = (const float*)d_in[31];
  const float* n0m1w = (const float*)d_in[32];
  const float* n0m1b = (const float*)d_in[33];
  const float* n0m2w = (const float*)d_in[34];
  const float* n0m2b = (const float*)d_in[35];
  const float* n1m1w = (const float*)d_in[36];
  const float* n1m1b = (const float*)d_in[37];
  const float* n1m2w = (const float*)d_in[38];
  const float* n1m2b = (const float*)d_in[39];
  float* out = (float*)d_out;

  // workspace layout (floats; every size is a multiple of 8 -> 32B alignment)
  float* w = (float*)d_ws;
  size_t o = 0;
  float* xs_out = w + o; o += (size_t)BN_ * LF_;     // transformer output
  float* xe     = w + o; o += (size_t)BN_ * DIN_;
  float* loce   = w + o; o += (size_t)BN_ * DLOC_;
  float* grw    = w + o; o += (size_t)BN_ * G_;
  float* gf     = w + o; o += (size_t)B_ * G_ * DIL_;
  float* eattr  = w + o; o += (size_t)B_ * EG_ * DE_;
  float* msgG   = w + o; o += (size_t)B_ * EG_ * DG_;
  float* aggG   = w + o; o += (size_t)B_ * G_ * DG_;
  float* hg1    = w + o; o += (size_t)B_ * G_ * DG_;
  float* gf2    = w + o; o += (size_t)B_ * G_ * DG_;
  float* newx   = w + o; o += (size_t)BN_ * (DIN_ + DG_);
  float* hW0    = w + o; o += (size_t)BN_ * DG_;
  float* aggN0  = w + o; o += (size_t)BN_ * DG_;
  float* cat0   = w + o; o += (size_t)BN_ * (DIN_ + DG_ + DG_);
  float* h0     = w + o; o += (size_t)BN_ * DG_;
  float* hW1    = w + o; o += (size_t)BN_ * DG_;
  float* aggN1  = w + o; o += (size_t)BN_ * DG_;
  float* cat1   = w + o; o += (size_t)BN_ * (DG_ + DG_);
  // packed f16 weight fragments
  _Float16* hp = (_Float16*)(w + o);
  size_t ho = 0;
  _Float16* pWin  = hp + ho; ho += (size_t)LF_ * 64;   // K=2688
  _Float16* pN0m1 = hp + ho; ho += (size_t)128 * 64;   // K=128 (rows 0..127 of n0m1w)
  _Float16* pN0m2 = hp + ho; ho += (size_t)192 * 64;   // K=192
  _Float16* pN1m1 = hp + ho; ho += (size_t)64  * 64;   // K=64  (rows 0..63 of n1m1w)
  _Float16* pN1m2 = hp + ho; ho += (size_t)128 * 64;   // K=128

  const int T = 256;

  // 0) pack all weight matrices into WMMA fragment order (weights-only deps)
  launch_pack(win,   LF_, pWin,  stream);
  launch_pack(n0m1w, 128, pN0m1, stream);
  launch_pack(n0m2w, 192, pN0m2, stream);
  launch_pack(n1m1w, 64,  pN1m1, stream);
  launch_pack(n1m2w, 128, pN1m2, stream);

  // 1) transformer per station
  tfm_kernel<<<BN_, T, 0, stream>>>(x, wqkv, bqkv, wo, bo, ln1g, ln1b,
                                    w1, b1, w2, b2, ln2g, ln2b, xs_out);
  // 2) xe = xs_flat @ win + b_in   (WMMA, K=2688)
  launch_wmma(xs_out, pWin, b_in, xe, LF_, LF_, 0, stream);
  // 3) loc embedding + soft grouping
  loc_grp_kernel<<<cdiv(BN_, T), T, 0, stream>>>(src_locs, wloc, bloc, wgrp, bgrp, loce, grw);
  // 4) group features gf = grw^T @ [xe|loce]
  gf_kernel<<<cdiv(B_ * G_ * DIL_, T), T, 0, stream>>>(grw, xe, loce, gf);
  // 5) group edge attributes
  eattr_kernel<<<cdiv(B_ * EG_ * DE_, T), T, 0, stream>>>(gf, wedge, bedge, eattr);
  // 6-8) group-graph layer 0
  gmsg_kernel<<<cdiv(B_ * EG_ * DG_, T), T, 0, stream>>>(gf, DIL_, eattr, g0m1w, g0m1b, msgG);
  gagg_kernel<<<cdiv(B_ * G_ * DG_, T), T, 0, stream>>>(msgG, aggG);
  gupd_kernel<<<cdiv(B_ * G_ * DG_, T), T, 0, stream>>>(gf, DIL_, aggG, g0m2w, g0m2b, hg1);
  // 9-11) group-graph layer 1
  gmsg_kernel<<<cdiv(B_ * EG_ * DG_, T), T, 0, stream>>>(hg1, DG_, eattr, g1m1w, g1m1b, msgG);
  gagg_kernel<<<cdiv(B_ * G_ * DG_, T), T, 0, stream>>>(msgG, aggG);
  gupd_kernel<<<cdiv(B_ * G_ * DG_, T), T, 0, stream>>>(hg1, DG_, aggG, g1m2w, g1m2b, gf2);
  // 12) new_x = [xe | grw @ gf2]
  newx_kernel<<<cdiv(BN_ * (DIN_ + DG_), T), T, 0, stream>>>(xe, grw, gf2, newx);
  // 13) hW0 = new_x @ n0m1w[0:128]   (WMMA, K=128)
  launch_wmma(newx, pN0m1, nullptr, hW0, 128, 128, 0, stream);
  // 14) station-graph agg layer 0 (dist folded in analytically)
  nagg_kernel<<<cdiv(BN_ * DG_, T), T, 0, stream>>>(hW0, n0m1w + (size_t)(DIN_ + DG_) * DG_,
                                                    n0m1b, src_locs, aggN0);
  // 15) cat0 = [new_x | aggN0]; 16) h0 = relu(cat0 @ n0m2w + b)  (WMMA, K=192)
  concat_kernel<<<cdiv(BN_ * 192, T), T, 0, stream>>>(newx, DIN_ + DG_, aggN0, DG_, cat0);
  launch_wmma(cat0, pN0m2, n0m2b, h0, 192, 192, 1, stream);
  // 17) hW1 = h0 @ n1m1w[0:64]   (WMMA, K=64)
  launch_wmma(h0, pN1m1, nullptr, hW1, 64, 64, 0, stream);
  // 18) station-graph agg layer 1
  nagg_kernel<<<cdiv(BN_ * DG_, T), T, 0, stream>>>(hW1, n1m1w + (size_t)DG_ * DG_,
                                                    n1m1b, src_locs, aggN1);
  // 19) cat1 = [h0 | aggN1]; 20) out = relu(cat1 @ n1m2w + b)  (WMMA, K=128)
  concat_kernel<<<cdiv(BN_ * 128, T), T, 0, stream>>>(h0, DG_, aggN1, DG_, cat1);
  launch_wmma(cat1, pN1m2, n1m2b, out, 128, 128, 1, stream);
}
